// ConvModule_68401649156298
// MI455X (gfx1250) — matrix operand order repaired and checked by reading, not compiled, not run on hardware
//
#include <hip/hip_runtime.h>
#include <hip/hip_bf16.h>

typedef __attribute__((ext_vector_type(16))) __bf16 v16bf;
typedef __attribute__((ext_vector_type(8)))  float  v8f;
typedef unsigned int u32;
typedef __attribute__((ext_vector_type(4)))  u32    u32x4;
typedef int int4n __attribute__((vector_size(16)));   // matches builtin param type

#define B_   16
#define T_   2048
#define D_   512
#define I_   1024
#define K_   31
#define M_   (B_ * T_)        // 32768 rows

// ---- CDNA5 async global->LDS path (guarded; falls back to sync copy) ------
#if defined(__has_builtin)
# if __has_builtin(__builtin_amdgcn_global_load_async_to_lds_b128)
#  define HAVE_ASYNC_LDS 1
# endif
#endif
#ifndef HAVE_ASYNC_LDS
# define HAVE_ASYNC_LDS 0
#endif

__device__ __forceinline__ void stage_b128(__bf16* ldst, const __bf16* gsrc) {
#if HAVE_ASYNC_LDS
    __builtin_amdgcn_global_load_async_to_lds_b128(
        (__attribute__((address_space(1))) int4n*)gsrc,
        (__attribute__((address_space(3))) int4n*)ldst, 0, 0);
#else
    *(u32x4*)ldst = *(const u32x4*)gsrc;
#endif
}

__device__ __forceinline__ void wait_async() {
#if HAVE_ASYNC_LDS
# if __has_builtin(__builtin_amdgcn_s_wait_asynccnt)
    __builtin_amdgcn_s_wait_asynccnt(0);
# else
    asm volatile("s_wait_asynccnt 0x0" ::: "memory");
# endif
#endif
}

// Load a 16-half (32B) WMMA fragment from two 16B chunks.
__device__ __forceinline__ v16bf load_frag(const __bf16* p0, const __bf16* p1) {
    union { v16bf v; u32x4 u[2]; } f;
    f.u[0] = *(const u32x4*)p0;
    f.u[1] = *(const u32x4*)p1;
    return f.v;
}

__device__ __forceinline__ v8f wmma_bf16(v16bf a, v16bf b, v8f c) {
    return __builtin_amdgcn_wmma_f32_16x16x32_bf16(false, a, false, b,
                                                   (short)0, c, false, false);
}

// ---------------------------------------------------------------------------
// Kernel 0: convert w1 [2I,D] and w2 [D,I] fp32 -> bf16 workspace copies
// ---------------------------------------------------------------------------
__global__ __launch_bounds__(256) void k0_cvt(const float* __restrict__ w1,
                                              const float* __restrict__ w2,
                                              __bf16* __restrict__ w1b,
                                              __bf16* __restrict__ w2b) {
    long i = (long)blockIdx.x * 256 + threadIdx.x;
    const long n1 = (long)(2 * I_) * D_;
    const long n2 = (long)D_ * I_;
    if (i < n1) w1b[i] = (__bf16)w1[i];
    if (i < n2) w2b[i] = (__bf16)w2[i];
}

// ---------------------------------------------------------------------------
// Kernel 1: LayerNorm + GEMM(D -> 2I) + SwiGLU  ->  u [M, I] bf16
// Block: 256 thr (8 waves). Tile: 32 rows x 128 swiglu channels.
// Each wave: 2 M-tiles x (act + gate) = 4 WMMA accumulators sharing B frags.
// Grid: (I/128, M/32)
// ---------------------------------------------------------------------------
__global__ __launch_bounds__(256) void k1_ln_gemm_swiglu(
    const float* __restrict__ x,
    const float* __restrict__ ln_g, const float* __restrict__ ln_b,
    const __bf16* __restrict__ w1b, const float* __restrict__ b1,
    __bf16* __restrict__ u)
{
    const int  nblk  = blockIdx.x;            // 0..7
    const long mbase = (long)blockIdx.y * 32; // row tile base
    const int  tid   = threadIdx.x;

    __shared__ __bf16 a_s[32][D_ + 8];        // padded rows: conflict-free frags
    __shared__ float  red[2][16][16];

    // ---- LayerNorm: two passes of 16 rows; 16 threads/row x 32 cols ----
    const int row16 = tid >> 4;
    const int seg   = tid & 15;
    for (int p = 0; p < 2; ++p) {
        const int row = row16 + p * 16;
        const float* xr = x + (mbase + row) * D_ + seg * 32;
        float vals[32];
        float s = 0.f, s2 = 0.f;
        #pragma unroll
        for (int j = 0; j < 32; ++j) { float v = xr[j]; vals[j] = v; s += v; s2 += v * v; }
        red[0][row16][seg] = s;
        red[1][row16][seg] = s2;
        __syncthreads();
        float rs = 0.f, rs2 = 0.f;
        #pragma unroll
        for (int j = 0; j < 16; ++j) { rs += red[0][row16][j]; rs2 += red[1][row16][j]; }
        const float mu   = rs * (1.f / D_);
        const float var  = rs2 * (1.f / D_) - mu * mu;
        const float rstd = __frsqrt_rn(var + 1e-5f);
        #pragma unroll
        for (int j = 0; j < 32; ++j) {
            const int d = seg * 32 + j;
            a_s[row][d] = (__bf16)((vals[j] - mu) * rstd * ln_g[d] + ln_b[d]);
        }
        __syncthreads();
    }

    // ---- WMMA GEMM: wave -> 16 swiglu channels, 2 M-tiles, act+gate ----
    const int wave = tid >> 5;
    const int lane = tid & 31;
    const int ln15 = lane & 15;
    const int hi   = lane >> 4;

    const int  ca = nblk * 128 + wave * 16 + ln15;  // act channel (w1 row, C col)
    const int  cg = I_ + ca;                        // gate channel
    const __bf16* wa    = w1b + (long)ca * D_;
    const __bf16* wg    = w1b + (long)cg * D_;
    const __bf16* arow0 = &a_s[ln15][0];
    const __bf16* arow1 = &a_s[16 + ln15][0];

    v8f acc_a0 = {}, acc_a1 = {}, acc_g0 = {}, acc_g1 = {};
    for (int k0 = 0; k0 < D_; k0 += 32) {
        const __bf16* ap0 = arow0 + k0 + hi * 8;
        const __bf16* ap1 = arow1 + k0 + hi * 8;
        v16bf af0 = load_frag(ap0, ap0 + 16);
        v16bf af1 = load_frag(ap1, ap1 + 16);
        const __bf16* bpa = wa + k0 + hi * 8;
        const __bf16* bpg = wg + k0 + hi * 8;
        v16bf bfa = load_frag(bpa, bpa + 16);
        v16bf bfg = load_frag(bpg, bpg + 16);
        acc_a0 = wmma_bf16(af0, bfa, acc_a0);
        acc_a1 = wmma_bf16(af1, bfa, acc_a1);
        acc_g0 = wmma_bf16(af0, bfg, acc_g0);
        acc_g1 = wmma_bf16(af1, bfg, acc_g1);
    }

    // ---- SwiGLU epilogue ----
    const float ba = b1[ca];
    const float bg = b1[cg];
    #pragma unroll
    for (int mt = 0; mt < 2; ++mt) {
        const v8f va = mt ? acc_a1 : acc_a0;
        const v8f vg = mt ? acc_g1 : acc_g0;
        #pragma unroll
        for (int r = 0; r < 8; ++r) {
            const float aa = va[r] + ba;
            const float gg = vg[r] + bg;
            const float si = gg * __frcp_rn(1.f + __expf(-gg));   // silu(g)
            const long  m  = mbase + mt * 16 + r + hi * 8;
            u[m * (long)I_ + ca] = (__bf16)(aa * si);
        }
    }
}

// ---------------------------------------------------------------------------
// Kernel 2: depthwise conv(K=31) + PReLU + GEMM(I -> D) + bias -> out fp32
// Block: 256 thr (8 waves). Tile: 32 rows x all 512 d-cols.
// u halo double-buffered in LDS via async global->LDS loads.
// Grid: (M/32)
// ---------------------------------------------------------------------------
__global__ __launch_bounds__(256) void k2_dwconv_prelu_gemm(
    const __bf16* __restrict__ u,
    const float* __restrict__ dw, const float* __restrict__ dwb,
    const float* __restrict__ alpha,
    const __bf16* __restrict__ w2b, const float* __restrict__ b2,
    float* __restrict__ out)
{
    const long bt0 = (long)blockIdx.x * 32;
    const int  t0  = (int)(bt0 % T_);        // tiles never cross batch boundary
    const int  tid = threadIdx.x;

    __shared__ __bf16 ust[2][62][64];        // halo rows t0-15 .. t0+46, x2 bufs
    __shared__ __bf16 vtile[32][64 + 8];     // A tile (post dwconv+PReLU), padded

    const int wave = tid >> 5;
    const int lane = tid & 31;
    const int ln15 = lane & 15;
    const int hi   = lane >> 4;

    const int ch = tid & 63;                 // this thread's channel in chunk
    const int rg = (tid >> 6) * 8;           // this thread's 8 output rows

    v8f acc[2][4] = {};

    auto stage = [&](int buf, int c0) {
        for (int idx = tid; idx < 62 * 8; idx += 256) {
            const int r  = idx >> 3;
            const int sg = idx & 7;
            const int t  = t0 - 15 + r;
            __bf16* dst = &ust[buf][r][sg * 8];
            if (t >= 0 && t < T_) {
                stage_b128(dst, u + (bt0 - t0 + t) * (long)I_ + c0 + sg * 8);
            } else {
                u32x4 z = {};
                *(u32x4*)dst = z;
            }
        }
    };

    stage(0, 0);                             // prologue: first halo chunk
    int buf = 0;

    for (int c0 = 0; c0 < I_; c0 += 64) {
        // per-thread channel params (independent of LDS state)
        const int c = c0 + ch;
        float taps[K_];
        #pragma unroll
        for (int k = 0; k < K_; ++k) taps[k] = dw[c * K_ + k];
        const float bias = dwb[c];
        const float al   = alpha[c];

        wait_async();
        __syncthreads();                     // ust[buf] ready; vtile reads done

        if (c0 + 64 < I_) stage(buf ^ 1, c0 + 64);   // overlap next halo

        // depthwise conv: sliding register window + PReLU -> vtile (bf16)
        float uv[38];
        #pragma unroll
        for (int j = 0; j < 38; ++j) uv[j] = (float)ust[buf][rg + j][ch];
        #pragma unroll
        for (int i = 0; i < 8; ++i) {
            float a = bias;
            #pragma unroll
            for (int k = 0; k < K_; ++k) a = fmaf(uv[i + k], taps[k], a);
            const float pv = a > 0.f ? a : al * a;
            vtile[rg + i][ch] = (__bf16)pv;
        }
        __syncthreads();                     // vtile ready

        // WMMA: 2 M-tiles x 4 N-tiles per wave share B fragments
        #pragma unroll
        for (int k2 = 0; k2 < 64; k2 += 32) {
            const __bf16* ap0 = &vtile[ln15][0] + k2 + hi * 8;
            const __bf16* ap1 = &vtile[16 + ln15][0] + k2 + hi * 8;
            v16bf af0 = load_frag(ap0, ap0 + 16);
            v16bf af1 = load_frag(ap1, ap1 + 16);
            #pragma unroll
            for (int p = 0; p < 4; ++p) {
                const int dcol = p * 128 + wave * 16 + ln15;
                const __bf16* bp = w2b + (long)dcol * I_ + c0 + k2 + hi * 8;
                v16bf bf = load_frag(bp, bp + 16);
                acc[0][p] = wmma_bf16(af0, bf, acc[0][p]);
                acc[1][p] = wmma_bf16(af1, bf, acc[1][p]);
            }
        }
        buf ^= 1;
    }

    // epilogue: + b2, store fp32
    #pragma unroll
    for (int mt = 0; mt < 2; ++mt) {
        #pragma unroll
        for (int p = 0; p < 4; ++p) {
            const int   d  = p * 128 + wave * 16 + ln15;
            const float bb = b2[d];
            #pragma unroll
            for (int r = 0; r < 8; ++r) {
                const long m = bt0 + mt * 16 + r + hi * 8;
                out[m * (long)D_ + d] = acc[mt][p][r] + bb;
            }
        }
    }
}

// ---------------------------------------------------------------------------
extern "C" void kernel_launch(void* const* d_in, const int* in_sizes, int n_in,
                              void* d_out, int out_size, void* d_ws, size_t ws_size,
                              hipStream_t stream) {
    const float* x     = (const float*)d_in[0];
    const float* ln_g  = (const float*)d_in[1];
    const float* ln_b  = (const float*)d_in[2];
    const float* w1    = (const float*)d_in[3];
    const float* b1    = (const float*)d_in[4];
    const float* dw    = (const float*)d_in[5];
    const float* dwb   = (const float*)d_in[6];
    const float* alpha = (const float*)d_in[7];
    const float* w2    = (const float*)d_in[8];
    const float* b2    = (const float*)d_in[9];
    float* out = (float*)d_out;

    // workspace layout (bytes): w1b [2MB] | w2b [1MB] | u [64MB]
    char* ws = (char*)d_ws;
    __bf16* w1b = (__bf16*)ws;
    __bf16* w2b = (__bf16*)(ws + (size_t)(2 * I_) * D_ * 2);
    __bf16* ub  = (__bf16*)(ws + (size_t)(2 * I_) * D_ * 2 + (size_t)D_ * I_ * 2);

    {
        const long n1 = (long)(2 * I_) * D_;
        dim3 grid((unsigned)((n1 + 255) / 256));
        k0_cvt<<<grid, 256, 0, stream>>>(w1, w2, w1b, w2b);
    }
    {
        dim3 grid(I_ / 128, M_ / 32);
        k1_ln_gemm_swiglu<<<grid, 256, 0, stream>>>(x, ln_g, ln_b, w1b, b1, ub);
    }
    {
        dim3 grid(M_ / 32);
        k2_dwconv_prelu_gemm<<<grid, 256, 0, stream>>>(ub, dw, dwb, alpha, w2b, b2, out);
    }
}
